// PointNet2Classifier_37288906064655
// MI455X (gfx1250) — matrix operand order repaired and checked by reading, not compiled, not run on hardware
//
#include <hip/hip_runtime.h>

// ---------------------------------------------------------------------------
// PointNet++ classifier forward for gfx1250 (MI455X).
// Fused per-group SA-MLP kernels using v_wmma_f32_16x16x32_bf16, LDS-resident
// activations, ds-atomic max pooling. FPS/ball-query as wave32 kernels.
// ---------------------------------------------------------------------------

typedef __bf16 bf16_t;
typedef __attribute__((ext_vector_type(16))) __bf16 v16bf;
typedef __attribute__((ext_vector_type(8)))  float  v8f;

struct LayerP { const bf16_t* W; const float* scale; const float* shift; int Cout; int CinPad; };

union AFrag { v16bf v; bf16_t e[16]; };
union CFrag { v8f  v; float  e[8];  };

// A-matrix 16x32 bf16 fragment from LDS (row-major X[Kp][stride]).
// Layout per ISA 7.12.2: lanes 0-15 M=lane, VGPR p<4 -> K=2p,2p+1 ; p>=4 -> K=16+2(p-4)..
// lanes 16-31 same M, K offset by 8 (p<4) / 24 (p>=4).
__device__ __forceinline__ AFrag load_a_lds(const bf16_t* X, int stride, int m0, int k0, int lane) {
  AFrag a;
  int l = lane & 15, hi = lane >> 4;
  const bf16_t* row = X + (size_t)(m0 + l) * stride + k0;
#pragma unroll
  for (int p = 0; p < 8; ++p) {
    int kb = (p < 4) ? ((hi ? 8 : 0) + 2 * p) : ((hi ? 24 : 16) + 2 * (p - 4));
    a.e[2 * p]     = row[kb];
    a.e[2 * p + 1] = row[kb + 1];
  }
  return a;
}

// B-matrix 32x16 bf16 fragment. B[k][n] = W[n][k] (W row-major [Cout][CinPad]).
// lanes 0-15: N=lane, K=k0+0..7,16..23 ; lanes 16-31: N=lane-16, K=k0+8..15,24..31.
__device__ __forceinline__ AFrag load_b_gl(const bf16_t* W, int cinPad, int n0, int k0, int lane) {
  AFrag b;
  int l = lane & 15, hi = lane >> 4;
  const bf16_t* src = W + (size_t)(n0 + l) * cinPad + k0 + (hi ? 8 : 0);
#pragma unroll
  for (int e = 0; e < 16; ++e) b.e[e] = src[(e < 8) ? e : e + 8];
  return b;
}

__device__ __forceinline__ void wmma_acc(CFrag& c, const AFrag& a, const AFrag& b) {
  c.v = __builtin_amdgcn_wmma_f32_16x16x32_bf16(false, a.v, false, b.v, (short)0, c.v, false, false);
}

// ---------------------------------------------------------------------------
// Input transpose: x (B,6,N) -> xyz (B,N,3), feats (B,N,3)
// ---------------------------------------------------------------------------
__global__ void k_transpose(const float* __restrict__ x, float* __restrict__ xyz,
                            float* __restrict__ feat) {
  int t = blockIdx.x * blockDim.x + threadIdx.x;
  if (t >= 16 * 4096) return;
  int b = t >> 12, n = t & 4095;
  const float* xb = x + (size_t)b * 6 * 4096;
#pragma unroll
  for (int c = 0; c < 3; ++c) xyz[(size_t)t * 3 + c]  = xb[(size_t)c * 4096 + n];
#pragma unroll
  for (int c = 0; c < 3; ++c) feat[(size_t)t * 3 + c] = xb[(size_t)(3 + c) * 4096 + n];
}

// ---------------------------------------------------------------------------
// Fold BN into scale/shift; convert & zero-pad weights to bf16 [Cout][CinPad]
// ---------------------------------------------------------------------------
__global__ void k_prep(const float* __restrict__ W, const float* __restrict__ bias,
                       const float* __restrict__ g, const float* __restrict__ be,
                       const float* __restrict__ m, const float* __restrict__ v,
                       int Cout, int Cin, int CinPad,
                       bf16_t* __restrict__ Wbf, float* __restrict__ scale,
                       float* __restrict__ shift) {
  int t = blockIdx.x * blockDim.x + threadIdx.x;
  if (t < Cout * CinPad) {
    int oc = t / CinPad, c = t % CinPad;
    Wbf[t] = (bf16_t)((c < Cin) ? W[(size_t)oc * Cin + c] : 0.0f);
  }
  if (t < Cout) {
    float sc = g[t] * rsqrtf(v[t] + 1e-5f);
    scale[t] = sc;
    shift[t] = (bias[t] - m[t]) * sc + be[t];
  }
}

// ---------------------------------------------------------------------------
// Farthest point sampling. One workgroup per batch, points + dist in LDS.
// ---------------------------------------------------------------------------
__global__ __launch_bounds__(256) void k_fps(const float* __restrict__ xyz, int N, int npoint,
                                             int* __restrict__ fidx) {
  extern __shared__ char smem[];
  float* px = (float*)smem;
  float* py = px + N;
  float* pz = py + N;
  float* dist = pz + N;
  float* rv = dist + N;
  int*   ri = (int*)(rv + 256);

  int b = blockIdx.x, tid = threadIdx.x;
  const float* src = xyz + (size_t)b * N * 3;
  for (int i = tid; i < N; i += 256) {
    px[i] = src[i * 3 + 0]; py[i] = src[i * 3 + 1]; pz[i] = src[i * 3 + 2];
    dist[i] = 1e10f;
  }
  __syncthreads();

  int far = 0;
  for (int it = 0; it < npoint; ++it) {
    if (tid == 0) fidx[b * npoint + it] = far;
    float cx = px[far], cy = py[far], cz = pz[far];
    float best = -1.0f; int bi = 0;
    for (int i = tid; i < N; i += 256) {
      float dx = px[i] - cx, dy = py[i] - cy, dz = pz[i] - cz;
      float d = dx * dx + dy * dy + dz * dz;
      float nd = fminf(dist[i], d);
      dist[i] = nd;
      if (nd > best) { best = nd; bi = i; }
    }
    rv[tid] = best; ri[tid] = bi;
    __syncthreads();
    for (int off = 128; off > 0; off >>= 1) {
      if (tid < off) {
        if (rv[tid + off] > rv[tid]) { rv[tid] = rv[tid + off]; ri[tid] = ri[tid + off]; }
      }
      __syncthreads();
    }
    far = ri[0];
    __syncthreads();
  }
}

// ---------------------------------------------------------------------------
// Gather sampled centroids: new_xyz[b][s] = xyz[b][fidx[b][s]]
// ---------------------------------------------------------------------------
__global__ void k_gather_xyz(const float* __restrict__ xyz, const int* __restrict__ fidx,
                             int N, int S, float* __restrict__ out) {
  int t = blockIdx.x * blockDim.x + threadIdx.x;
  if (t >= 16 * S) return;
  int b = t / S;
  int pi = fidx[t];
  const float* p = xyz + ((size_t)b * N + pi) * 3;
  out[(size_t)t * 3 + 0] = p[0];
  out[(size_t)t * 3 + 1] = p[1];
  out[(size_t)t * 3 + 2] = p[2];
}

// ---------------------------------------------------------------------------
// Ball query: one wave32 per query. Ordered compaction of first K in-radius
// indices (matches sort-then-truncate of reference); pad with first index.
// ---------------------------------------------------------------------------
__global__ __launch_bounds__(256) void k_ballquery(const float* __restrict__ xyz,
                                                   const float* __restrict__ new_xyz,
                                                   int N, int S, int K, float r2,
                                                   int* __restrict__ idx) {
  int gw = (blockIdx.x * blockDim.x + threadIdx.x) >> 5;
  int lane = threadIdx.x & 31;
  if (gw >= 16 * S) return;
  const float* q = new_xyz + (size_t)gw * 3;
  float cx = q[0], cy = q[1], cz = q[2];
  const float* pts = xyz + (size_t)(gw / S) * N * 3;
  int* out = idx + (size_t)gw * K;

  int written = 0, first = -1;
  for (int base = 0; base < N; base += 32) {
    int i = base + lane;
    float dx = pts[i * 3 + 0] - cx, dy = pts[i * 3 + 1] - cy, dz = pts[i * 3 + 2] - cz;
    bool ok = (dx * dx + dy * dy + dz * dz) <= r2;
    unsigned mm = (unsigned)__ballot(ok);
    if (first < 0 && mm) first = base + (__ffs(mm) - 1);
    int pre = __popc(mm & ((1u << lane) - 1u));
    if (ok) {
      int p = written + pre;
      if (p < K) out[p] = i;
    }
    written += __popc(mm);
    if (written >= K) break;
  }
  if (written < K) {
    int f = (first < 0) ? 0 : first;
    for (int j = written + lane; j < K; j += 32) out[j] = f;
  }
}

// ---------------------------------------------------------------------------
// Fused SA group MLP + max pooling. One workgroup (8 waves) per group (b,s).
// Gathers K x Cin tile into LDS (bf16), runs up to 3 WMMA layers (BN+ReLU
// folded), max-pools final layer over K via ds atomic max (relu => >=0).
// ---------------------------------------------------------------------------
__global__ __launch_bounds__(256) void k_sa_mlp(
    const float* __restrict__ xyz_src, const float* __restrict__ pts_src,
    int Cfeat, int Nsrc, const float* __restrict__ new_xyz, const int* __restrict__ idx,
    int S, int Kp, int Cin, LayerP L0, LayerP L1, LayerP L2, int nlayers,
    int bufElems, float* __restrict__ out) {
  extern __shared__ char smem[];
  bf16_t* bufA = (bf16_t*)smem;
  bf16_t* bufB = bufA + bufElems;
  unsigned* colmax = (unsigned*)(bufB + bufElems);

  int group = blockIdx.x;
  int b = group / S;
  int tid = threadIdx.x, lane = tid & 31, wave = tid >> 5;

  // ---- gather grouped features: [Kp][C0] bf16, cols = [xyz-rel, feats, pad0]
  const int* gi = idx + (size_t)group * Kp;
  float nx = new_xyz[(size_t)group * 3 + 0];
  float ny = new_xyz[(size_t)group * 3 + 1];
  float nz = new_xyz[(size_t)group * 3 + 2];
  int C0 = L0.CinPad;
  for (int t = tid; t < Kp * C0; t += 256) {
    int kp = t / C0, c = t % C0;
    float val = 0.0f;
    int pi = gi[kp];
    if (c < 3)
      val = xyz_src[((size_t)b * Nsrc + pi) * 3 + c] - (c == 0 ? nx : (c == 1 ? ny : nz));
    else if (c < Cin)
      val = pts_src[((size_t)b * Nsrc + pi) * Cfeat + (c - 3)];
    bufA[t] = (bf16_t)val;
  }
  __syncthreads();

  // ---- hidden layers (ping-pong LDS buffers)
  const bf16_t* X = bufA;
  for (int li = 0; li < nlayers - 1; ++li) {
    LayerP L = (li == 0) ? L0 : L1;
    bf16_t* Y = (li == 0) ? bufB : bufA;
    int nTN = L.Cout >> 4, nTM = Kp >> 4, nT = nTN * nTM;
    for (int t = wave; t < nT; t += 8) {
      int m0 = (t / nTN) << 4, n0 = (t % nTN) << 4;
      CFrag c;
#pragma unroll
      for (int e = 0; e < 8; ++e) c.e[e] = 0.0f;
      for (int k0 = 0; k0 < L.CinPad; k0 += 32) {
        AFrag a  = load_a_lds(X, L.CinPad, m0, k0, lane);
        AFrag bb = load_b_gl(L.W, L.CinPad, n0, k0, lane);
        wmma_acc(c, a, bb);
      }
      int l = lane & 15, hi = lane >> 4, col = n0 + l;
      float sc = L.scale[col], sh = L.shift[col];
#pragma unroll
      for (int r = 0; r < 8; ++r) {
        int row = m0 + r + (hi ? 8 : 0);
        float y = c.e[r] * sc + sh;
        y = (y > 0.0f) ? y : 0.0f;
        Y[(size_t)row * L.Cout + col] = (bf16_t)y;
      }
    }
    __syncthreads();
    X = Y;
  }

  // ---- final layer fused with max-over-K
  LayerP LF = (nlayers == 3) ? L2 : L1;
  for (int c = tid; c < LF.Cout; c += 256) colmax[c] = 0u;
  __syncthreads();
  int nTN = LF.Cout >> 4, nTM = Kp >> 4;
  int l = lane & 15, hi = lane >> 4;
  for (int nt = wave; nt < nTN; nt += 8) {
    int n0 = nt << 4, col = n0 + l;
    float sc = LF.scale[col], sh = LF.shift[col];
    float mx = 0.0f;
    for (int mt = 0; mt < nTM; ++mt) {
      CFrag c;
#pragma unroll
      for (int e = 0; e < 8; ++e) c.e[e] = 0.0f;
      for (int k0 = 0; k0 < LF.CinPad; k0 += 32) {
        AFrag a  = load_a_lds(X, LF.CinPad, mt << 4, k0, lane);
        AFrag bb = load_b_gl(LF.W, LF.CinPad, n0, k0, lane);
        wmma_acc(c, a, bb);
      }
#pragma unroll
      for (int r = 0; r < 8; ++r) {
        float y = c.e[r] * sc + sh;
        y = (y > 0.0f) ? y : 0.0f;
        if (y > mx) mx = y;
      }
    }
    atomicMax(&colmax[col], __float_as_uint(mx));  // relu => non-negative: bit order == float order
  }
  __syncthreads();
  float* o = out + (size_t)group * LF.Cout;
  for (int c = tid; c < LF.Cout; c += 256) o[c] = __uint_as_float(colmax[c]);
}

// ---------------------------------------------------------------------------
// Identity idx + zero centroids for the global SA stage.
// ---------------------------------------------------------------------------
__global__ void k_init_gsa(int* __restrict__ idxg, float* __restrict__ nxyzg) {
  int t = blockIdx.x * blockDim.x + threadIdx.x;
  if (t < 16 * 64) idxg[t] = t & 63;
  if (t < 48) nxyzg[t] = 0.0f;
}

// ---------------------------------------------------------------------------
// FC layer (16 rows) via WMMA. relu + folded BN via scale/shift.
// ---------------------------------------------------------------------------
__global__ __launch_bounds__(256) void k_fc(const float* __restrict__ Xg,
                                            const bf16_t* __restrict__ W,
                                            const float* __restrict__ scale,
                                            const float* __restrict__ shift,
                                            int Cin, int Cout, float* __restrict__ out) {
  extern __shared__ char smem[];
  bf16_t* X = (bf16_t*)smem;  // [16][Cin]
  int tid = threadIdx.x, lane = tid & 31, wave = tid >> 5;
  for (int t = tid; t < 16 * Cin; t += 256) X[t] = (bf16_t)Xg[t];
  __syncthreads();
  int nt = blockIdx.x * 8 + wave;
  if (nt < (Cout >> 4)) {
    int n0 = nt << 4, l = lane & 15, hi = lane >> 4, col = n0 + l;
    CFrag c;
#pragma unroll
    for (int e = 0; e < 8; ++e) c.e[e] = 0.0f;
    for (int k0 = 0; k0 < Cin; k0 += 32) {
      AFrag a  = load_a_lds(X, Cin, 0, k0, lane);
      AFrag bb = load_b_gl(W, Cin, n0, k0, lane);
      wmma_acc(c, a, bb);
    }
    float sc = scale[col], sh = shift[col];
#pragma unroll
    for (int r = 0; r < 8; ++r) {
      int row = r + (hi ? 8 : 0);
      float y = c.e[r] * sc + sh;
      out[(size_t)row * Cout + col] = (y > 0.0f) ? y : 0.0f;
    }
  }
}

// Tiny final linear layer (16x256 @ 256x2 + bias), scalar.
__global__ void k_fc3(const float* __restrict__ h, const float* __restrict__ W,
                      const float* __restrict__ b, float* __restrict__ out) {
  int t = blockIdx.x * blockDim.x + threadIdx.x;
  if (t >= 32) return;
  int row = t >> 1, o = t & 1;
  float acc = b[o];
  for (int c = 0; c < 256; ++c) acc += h[row * 256 + c] * W[o * 256 + c];
  out[row * 2 + o] = acc;
}

// ---------------------------------------------------------------------------
// Host launcher
// ---------------------------------------------------------------------------
extern "C" void kernel_launch(void* const* d_in, const int* in_sizes, int n_in,
                              void* d_out, int out_size, void* d_ws, size_t ws_size,
                              hipStream_t stream) {
  (void)in_sizes; (void)n_in; (void)out_size; (void)ws_size;
  const float* x = (const float*)d_in[0];
  // pytree (sorted-key) flat order: fc1(1-6) fc2(7-12) fc3(13-14) gsa(15-26)
  // sa1(27-44) sa2(45-62) sa3(63-80); each _mlp block: W*, b*, be*, g*, m*, v*
  const float* fc3W = (const float*)d_in[13];
  const float* fc3b = (const float*)d_in[14];

  char* ws = (char*)d_ws;
  size_t off = 0;
  auto alloc = [&](size_t bytes) -> void* {
    off = (off + 255) & ~(size_t)255;
    void* p = ws + off;
    off += bytes;
    return p;
  };

  float* xyz0  = (float*)alloc((size_t)16 * 4096 * 3 * 4);
  float* feat0 = (float*)alloc((size_t)16 * 4096 * 3 * 4);
  int*   fidx1 = (int*)alloc((size_t)16 * 1024 * 4);
  float* nxyz1 = (float*)alloc((size_t)16 * 1024 * 3 * 4);
  int*   idx1  = (int*)alloc((size_t)16 * 1024 * 32 * 4);
  float* p1    = (float*)alloc((size_t)16 * 1024 * 128 * 4);
  int*   fidx2 = (int*)alloc((size_t)16 * 256 * 4);
  float* nxyz2 = (float*)alloc((size_t)16 * 256 * 3 * 4);
  int*   idx2  = (int*)alloc((size_t)16 * 256 * 64 * 4);
  float* p2    = (float*)alloc((size_t)16 * 256 * 256 * 4);
  int*   fidx3 = (int*)alloc((size_t)16 * 64 * 4);
  float* nxyz3 = (float*)alloc((size_t)16 * 64 * 3 * 4);
  int*   idx3  = (int*)alloc((size_t)16 * 64 * 128 * 4);
  float* p3    = (float*)alloc((size_t)16 * 64 * 512 * 4);
  int*   idxg  = (int*)alloc((size_t)16 * 64 * 4);
  float* nxyzg = (float*)alloc((size_t)16 * 3 * 4);
  float* h1    = (float*)alloc((size_t)16 * 512 * 4);
  float* h2    = (float*)alloc((size_t)16 * 256 * 4);

  // 13 folded layers: sa1[0-2] sa2[3-5] sa3[6-8] gsa[9-10] fc1[11] fc2[12]
  struct LC { int Cout, Cin, CinPad, base, li, n; };
  const LC lc[13] = {
      {64, 6, 32, 27, 0, 3},    {64, 64, 64, 27, 1, 3},   {128, 64, 64, 27, 2, 3},
      {128, 131, 160, 45, 0, 3},{128, 128, 128, 45, 1, 3},{256, 128, 128, 45, 2, 3},
      {256, 259, 288, 63, 0, 3},{256, 256, 256, 63, 1, 3},{512, 256, 256, 63, 2, 3},
      {512, 515, 544, 15, 0, 2},{1024, 512, 512, 15, 1, 2},
      {512, 1024, 1024, 1, 0, 1},{256, 512, 512, 7, 0, 1}};
  bf16_t* Wf[13]; float* Sc[13]; float* Sh[13];
  for (int i = 0; i < 13; ++i) {
    Wf[i] = (bf16_t*)alloc((size_t)lc[i].Cout * lc[i].CinPad * 2);
    Sc[i] = (float*)alloc((size_t)lc[i].Cout * 4);
    Sh[i] = (float*)alloc((size_t)lc[i].Cout * 4);
  }
  for (int i = 0; i < 13; ++i) {
    const LC& c = lc[i];
    int total = c.Cout * c.CinPad;
    k_prep<<<(total + 255) / 256, 256, 0, stream>>>(
        (const float*)d_in[c.base + c.li],              // W
        (const float*)d_in[c.base + c.n + c.li],        // b
        (const float*)d_in[c.base + 3 * c.n + c.li],    // g
        (const float*)d_in[c.base + 2 * c.n + c.li],    // be
        (const float*)d_in[c.base + 4 * c.n + c.li],    // m
        (const float*)d_in[c.base + 5 * c.n + c.li],    // v
        c.Cout, c.Cin, c.CinPad, Wf[i], Sc[i], Sh[i]);
  }

  auto LP = [&](int i) { return LayerP{Wf[i], Sc[i], Sh[i], lc[i].Cout, lc[i].CinPad}; };

  k_transpose<<<(16 * 4096 + 255) / 256, 256, 0, stream>>>(x, xyz0, feat0);

  // ---- SA1: N=4096 -> S=1024, K=32, r=0.5
  k_fps<<<16, 256, (size_t)4 * 4096 * 4 + 256 * 8, stream>>>(xyz0, 4096, 1024, fidx1);
  k_gather_xyz<<<(16 * 1024 + 255) / 256, 256, 0, stream>>>(xyz0, fidx1, 4096, 1024, nxyz1);
  k_ballquery<<<(16 * 1024 * 32) / 256, 256, 0, stream>>>(xyz0, nxyz1, 4096, 1024, 32, 0.25f, idx1);
  {
    int Kp = 32, bufE = Kp * 64;  // maxC over {CinPad0=32, hidden Couts 64,64}
    size_t sm = (size_t)4 * bufE + 4 * 128;
    k_sa_mlp<<<16 * 1024, 256, sm, stream>>>(xyz0, feat0, 3, 4096, nxyz1, idx1,
                                             1024, Kp, 6, LP(0), LP(1), LP(2), 3, bufE, p1);
  }

  // ---- SA2: N=1024 -> S=256, K=64, r=1.0
  k_fps<<<16, 256, (size_t)4 * 1024 * 4 + 256 * 8, stream>>>(nxyz1, 1024, 256, fidx2);
  k_gather_xyz<<<(16 * 256 + 255) / 256, 256, 0, stream>>>(nxyz1, fidx2, 1024, 256, nxyz2);
  k_ballquery<<<(16 * 256 * 32) / 256, 256, 0, stream>>>(nxyz1, nxyz2, 1024, 256, 64, 1.0f, idx2);
  {
    int Kp = 64, bufE = Kp * 160;  // maxC {160,128,128}
    size_t sm = (size_t)4 * bufE + 4 * 256;
    k_sa_mlp<<<16 * 256, 256, sm, stream>>>(nxyz1, p1, 128, 1024, nxyz2, idx2,
                                            256, Kp, 131, LP(3), LP(4), LP(5), 3, bufE, p2);
  }

  // ---- SA3: N=256 -> S=64, K=128, r=2.0
  k_fps<<<16, 256, (size_t)4 * 256 * 4 + 256 * 8, stream>>>(nxyz2, 256, 64, fidx3);
  k_gather_xyz<<<(16 * 64 + 255) / 256, 256, 0, stream>>>(nxyz2, fidx3, 256, 64, nxyz3);
  k_ballquery<<<(16 * 64 * 32) / 256, 256, 0, stream>>>(nxyz2, nxyz3, 256, 64, 128, 4.0f, idx3);
  {
    int Kp = 128, bufE = Kp * 288;  // maxC {288,256,256}
    size_t sm = (size_t)4 * bufE + 4 * 512;
    k_sa_mlp<<<16 * 64, 256, sm, stream>>>(nxyz2, p2, 256, 256, nxyz3, idx3,
                                           64, Kp, 259, LP(6), LP(7), LP(8), 3, bufE, p3);
  }

  // ---- Global SA: all 64 pts per batch, identity idx, zero centroid
  k_init_gsa<<<(16 * 64 + 255) / 256, 256, 0, stream>>>(idxg, nxyzg);
  float* gfeat = (float*)d_out + 32;  // d_out = [logits(32) | gfeat(16x1024)]
  {
    int Kp = 64, bufE = Kp * 544;  // maxC {544,512}
    size_t sm = (size_t)4 * bufE + 4 * 1024;
    k_sa_mlp<<<16, 256, sm, stream>>>(nxyz3, p3, 512, 64, nxyzg, idxg,
                                      1, Kp, 515, LP(9), LP(10), LP(10), 2, bufE, gfeat);
  }

  // ---- FC head
  k_fc<<<(32 + 7) / 8, 256, (size_t)16 * 1024 * 2, stream>>>(gfeat, Wf[11], Sc[11], Sh[11], 1024, 512, h1);
  k_fc<<<(16 + 7) / 8, 256, (size_t)16 * 512 * 2, stream>>>(h1, Wf[12], Sc[12], Sh[12], 512, 256, h2);
  k_fc3<<<1, 32, 0, stream>>>(h2, fc3W, fc3b, (float*)d_out);
}
